// DecoderForLarge_72361609003097
// MI455X (gfx1250) — matrix operand order repaired and checked by reading, not compiled, not run on hardware
//
#include <hip/hip_runtime.h>
#include <hip/hip_bf16.h>
#include <math.h>
#include <stdint.h>

// ---------------- problem constants ----------------
#define NB   32
#define NN   2000
#define NG   200
#define HDIM 128

typedef __attribute__((ext_vector_type(16))) __bf16 v16bf;
typedef __attribute__((ext_vector_type(8)))  float  v8f;

// 1/(NH*sqrt(H)) = 1/(8*sqrt(128))
#define INV_SCALE   0.011048543456039805f
#define INV_SQRT2   0.70710678118654752f
#define A_LDK       136   // padded bf16 row stride (136*2B=272B -> conflict-free)

// ---------------- WMMA helpers ----------------
__device__ __forceinline__ v8f wmma_bf16(v16bf a, v16bf b, v8f c) {
  return __builtin_amdgcn_wmma_f32_16x16x32_bf16(false, a, false, b, (short)0, c,
                                                 false, false);
}

// A fragment: 16x32 bf16 tile, row-major with row stride ldk (bf16 units).
__device__ __forceinline__ v16bf load_a_frag(const __bf16* As, int ldk, int k0, int lane) {
  int m  = lane & 15;
  int hi = lane >> 4;
  const __bf16* p = As + m * ldk + k0 + hi * 16;
  v16bf a;
#pragma unroll
  for (int e = 0; e < 8; ++e) a[e] = p[e];
#pragma unroll
  for (int e = 0; e < 8; ++e) a[8 + e] = p[8 + e];
  return a;
}

// B fragment from a row-major (outcol, k) matrix: B[k][n] = M[n][k].
__device__ __forceinline__ v16bf load_b_frag(const __bf16* Mt, int ldk, int k0, int lane) {
  int n  = lane & 15;
  int kb = (lane >> 4) * 16;
  const __bf16* p = Mt + n * ldk + k0 + kb;
  v16bf b;
#pragma unroll
  for (int e = 0; e < 16; ++e) b[e] = p[e];
  return b;
}

// Async 16B global->LDS copy (CDNA5, tracked with ASYNCcnt). GV addressing.
__device__ __forceinline__ void async_copy16(uint32_t lds_off, const void* gaddr) {
  asm volatile("global_load_async_to_lds_b128 %0, %1, off"
               :: "v"(lds_off), "v"(gaddr) : "memory");
}
__device__ __forceinline__ void wait_async0() {
  asm volatile("s_wait_asynccnt 0x0" ::: "memory");
}

// ---------------- K1: Wsum = Wq_last + Wq_first ----------------
__global__ void k_wsum(const float* __restrict__ wl, const float* __restrict__ wf,
                       float* __restrict__ wsum) {
  int i = blockIdx.x * blockDim.x + threadIdx.x;
  if (i < HDIM * HDIM) wsum[i] = wl[i] + wf[i];
}

// ---------------- K2: q_graph[b] = mean_n(emb[b]) @ Wq_graph^T ----------------
__global__ void k_qgraph(const float* __restrict__ emb, const float* __restrict__ Wqg,
                         float* __restrict__ qg) {
  __shared__ float me[HDIM];
  int b = blockIdx.x, h = threadIdx.x;           // 128 threads
  const float* e = emb + (size_t)b * NN * HDIM;
  float s = 0.f;
  for (int n = 0; n < NN; ++n) s += e[(size_t)n * HDIM + h];
  me[h] = s * (1.0f / NN);
  __syncthreads();
  float acc = 0.f;
#pragma unroll 8
  for (int k = 0; k < HDIM; ++k) acc += me[k] * Wqg[h * HDIM + k];
  qg[(size_t)b * HDIM + h] = acc;
}

// ---------------- K3: Y(bf16)[M x128] = X(f32)[M x128] @ W(f32)[128x128]^T ------
__global__ void k_gemm_xwt(const float* __restrict__ X, const float* __restrict__ W,
                           __bf16* __restrict__ Y) {
  __shared__ __bf16 As[16][A_LDK];
  __shared__ __bf16 Ws[HDIM][A_LDK];
  int tid = threadIdx.x;                          // 256 threads = 8 waves
  int rowbase = blockIdx.x * 16;
  for (int i = tid; i < 16 * HDIM; i += 256) {
    int r = i >> 7, c = i & 127;
    As[r][c] = (__bf16)X[(size_t)(rowbase + r) * HDIM + c];
  }
  for (int i = tid; i < HDIM * HDIM; i += 256) {
    int r = i >> 7, c = i & 127;
    Ws[r][c] = (__bf16)W[i];
  }
  __syncthreads();
  int wave = tid >> 5, lane = tid & 31;
  int colbase = wave * 16;
  v8f acc = {};
#pragma unroll
  for (int k0 = 0; k0 < HDIM; k0 += 32) {
    v16bf a = load_a_frag(&As[0][0], A_LDK, k0, lane);
    v16bf b = load_b_frag(&Ws[colbase][0], A_LDK, k0, lane);
    acc = wmma_bf16(a, b, acc);
  }
  int n  = colbase + (lane & 15);
  int mo = (lane >> 4) * 8;
#pragma unroll
  for (int r = 0; r < 8; ++r)
    Y[(size_t)(rowbase + r + mo) * HDIM + n] = (__bf16)acc[r];
}

// ---------------- K4: sumvis[b,g,:] = sum_n vis[b,g,n] * emb[b,n,:] ------------
__global__ void k_sumvis(const float* __restrict__ emb, const float* __restrict__ mask,
                         float* __restrict__ sv) {
  __shared__ __bf16 EsT[HDIM][40];   // [h][k-in-chunk], chunk=32, pad->40
  __shared__ __bf16 As[16][40];      // [g][k-in-chunk] visited 0/1
  int b = blockIdx.y, gbase = blockIdx.x * 16;
  int tid = threadIdx.x;
  int wave = tid >> 5, lane = tid & 31, colbase = wave * 16;
  const float* embB  = emb  + (size_t)b * NN * HDIM;
  const float* maskB = mask + ((size_t)b * NG + gbase) * NN;
  v8f acc = {};
  for (int nc = 0; nc < NN; nc += 32) {
    for (int i = tid; i < 32 * HDIM; i += 256) {     // transpose-stage emb chunk
      int r = i >> 7, c = i & 127;
      int node = nc + r;
      float v = (node < NN) ? embB[(size_t)node * HDIM + c] : 0.0f;
      EsT[c][r] = (__bf16)v;
    }
    for (int i = tid; i < 16 * 32; i += 256) {       // stage 0/1 visited
      int r = i >> 5, c = i & 31;
      int g = gbase + r, node = nc + c;
      float v = 0.0f;
      if (g < NG && node < NN) {
        float mv = maskB[(size_t)r * NN + node];
        v = (mv < -1e30f) ? 1.0f : 0.0f;
      }
      As[r][c] = (__bf16)v;
    }
    __syncthreads();
    v16bf a  = load_a_frag(&As[0][0], 40, 0, lane);
    v16bf bb = load_b_frag(&EsT[colbase][0], 40, 0, lane);
    acc = wmma_bf16(a, bb, acc);
    __syncthreads();
  }
  int n  = colbase + (lane & 15);
  int mo = (lane >> 4) * 8;
#pragma unroll
  for (int r = 0; r < 8; ++r) {
    int g = gbase + r + mo;
    if (g < NG) sv[((size_t)b * NG + g) * HDIM + n] = acc[r];
  }
}

// ---- K5: final_q = gather(emb,last)@Wsum^T + (sumvis/N)@Wv^T + q_graph --------
__global__ void k_finalq(const float* __restrict__ emb, const int* __restrict__ last_node,
                         const float* __restrict__ sv, const float* __restrict__ Wsum,
                         const float* __restrict__ Wv, const float* __restrict__ qg,
                         float* __restrict__ fq) {
  __shared__ __bf16 A1[16][A_LDK], A2[16][A_LDK];
  __shared__ __bf16 W1[HDIM][A_LDK], W2[HDIM][A_LDK];
  int tid = threadIdx.x, rowbase = blockIdx.x * 16;   // rows flat over (b,g)
  for (int i = tid; i < 16 * HDIM; i += 256) {
    int r = i >> 7, c = i & 127;
    int row = rowbase + r;
    int b = row / NG, g = row % NG;
    int ln = last_node[(size_t)b * NG + g];
    A1[r][c] = (__bf16)emb[((size_t)b * NN + ln) * HDIM + c];
    A2[r][c] = (__bf16)(sv[(size_t)row * HDIM + c] * (1.0f / NN));
  }
  for (int i = tid; i < HDIM * HDIM; i += 256) {
    int r = i >> 7, c = i & 127;
    W1[r][c] = (__bf16)Wsum[i];
    W2[r][c] = (__bf16)Wv[i];
  }
  __syncthreads();
  int wave = tid >> 5, lane = tid & 31, colbase = wave * 16;
  v8f acc = {};
#pragma unroll
  for (int k0 = 0; k0 < HDIM; k0 += 32)
    acc = wmma_bf16(load_a_frag(&A1[0][0], A_LDK, k0, lane),
                    load_b_frag(&W1[colbase][0], A_LDK, k0, lane), acc);
#pragma unroll
  for (int k0 = 0; k0 < HDIM; k0 += 32)
    acc = wmma_bf16(load_a_frag(&A2[0][0], A_LDK, k0, lane),
                    load_b_frag(&W2[colbase][0], A_LDK, k0, lane), acc);
  int n  = colbase + (lane & 15);
  int mo = (lane >> 4) * 8;
#pragma unroll
  for (int r = 0; r < 8; ++r) {
    int row = rowbase + r + mo;
    int b = row / NG;
    fq[(size_t)row * HDIM + n] = acc[r] + qg[(size_t)b * HDIM + n];
  }
}

// ---- K6: scores + distance + tanh clip + mask (writes logits into d_out) ------
// A (fq) and B (K) tiles are both bf16 in ws -> staged into LDS via
// global_load_async_to_lds_b128 (CDNA5 async path, ASYNCcnt).
__global__ void k_scores(const __bf16* __restrict__ fqbf, const __bf16* __restrict__ Kbf,
                         const float* __restrict__ coords, const int* __restrict__ last_node,
                         const float* __restrict__ mask, float* __restrict__ out) {
  __shared__ __bf16 As[16][A_LDK];     //  4.4 KB
  __shared__ __bf16 Bs[128][A_LDK];    // 34.8 KB  (K rows for this 128-col chunk)
  __shared__ float lc[16][2];
  int b = blockIdx.z, gbase = blockIdx.y * 16, cbase0 = blockIdx.x * 128;
  int tid = threadIdx.x;

  const __bf16* KB  = Kbf  + (size_t)b * NN * HDIM;
  const __bf16* FQB = fqbf + (size_t)b * NG * HDIM;
  uint32_t as_base = (uint32_t)(uintptr_t)&As[0][0];
  uint32_t bs_base = (uint32_t)(uintptr_t)&Bs[0][0];

  // async-stage B tile: 128 rows x 256B, 16B chunks into 272B-padded LDS rows
  for (int i = tid; i < 128 * 16; i += 256) {
    int r = i >> 4, c = i & 15;
    int rr = cbase0 + r; if (rr > NN - 1) rr = NN - 1;   // clamp (dead cols skipped)
    async_copy16(bs_base + (uint32_t)r * (A_LDK * 2) + (uint32_t)c * 16,
                 KB + (size_t)rr * HDIM + c * 8);
  }
  // async-stage A tile: 16 rows x 256B = 256 chunks, one per thread
  {
    int r = tid >> 4, c = tid & 15;
    int g = gbase + r; if (g > NG - 1) g = NG - 1;       // clamp (dead rows skipped)
    async_copy16(as_base + (uint32_t)r * (A_LDK * 2) + (uint32_t)c * 16,
                 FQB + (size_t)g * HDIM + c * 8);
  }
  if (tid < 16) {
    int g  = gbase + tid;
    int ln = (g < NG) ? last_node[(size_t)b * NG + g] : 0;
    lc[tid][0] = coords[((size_t)b * NN + ln) * 2 + 0];
    lc[tid][1] = coords[((size_t)b * NN + ln) * 2 + 1];
  }
  wait_async0();
  __syncthreads();

  int wave = tid >> 5, lane = tid & 31;
  int colbase = cbase0 + wave * 16;
  if (colbase >= NN) return;                     // uniform per-wave exit
  v8f acc = {};
#pragma unroll
  for (int k0 = 0; k0 < HDIM; k0 += 32) {
    v16bf a  = load_a_frag(&As[0][0], A_LDK, k0, lane);
    v16bf bb = load_b_frag(&Bs[wave * 16][0], A_LDK, k0, lane);
    acc = wmma_bf16(a, bb, acc);
  }
  int n  = colbase + (lane & 15);
  int mo = (lane >> 4) * 8;
  float cx = coords[((size_t)b * NN + n) * 2 + 0];
  float cy = coords[((size_t)b * NN + n) * 2 + 1];
#pragma unroll
  for (int r = 0; r < 8; ++r) {
    int g = gbase + r + mo;
    if (g >= NG) continue;
    float dx = lc[r + mo][0] - cx, dy = lc[r + mo][1] - cy;
    float dist = sqrtf(dx * dx + dy * dy);
    float s = acc[r] * INV_SCALE - dist * INV_SQRT2;
    float t = 10.0f * tanhf(s);
    float mv = mask[((size_t)b * NG + g) * NN + n];
    float mp = (mv < -1e30f) ? -1e8f : mv;
    out[((size_t)b * NG + g) * NN + n] = t + mp;
  }
}

// ---------------- K7: in-place row softmax over N ----------------
__global__ void k_softmax(float* __restrict__ out) {
  __shared__ float red[256];
  float* row = out + (size_t)blockIdx.x * NN;
  int tid = threadIdx.x;
  float mx = -INFINITY;
  for (int n = tid; n < NN; n += 256) mx = fmaxf(mx, row[n]);
  red[tid] = mx;
  __syncthreads();
  for (int s = 128; s > 0; s >>= 1) {
    if (tid < s) red[tid] = fmaxf(red[tid], red[tid + s]);
    __syncthreads();
  }
  mx = red[0];
  __syncthreads();
  float sum = 0.f;
  for (int n = tid; n < NN; n += 256) sum += __expf(row[n] - mx);
  red[tid] = sum;
  __syncthreads();
  for (int s = 128; s > 0; s >>= 1) {
    if (tid < s) red[tid] += red[tid + s];
    __syncthreads();
  }
  float inv = 1.0f / red[0];
  for (int n = tid; n < NN; n += 256) row[n] = __expf(row[n] - mx) * inv;
}

// ---------------- host-side launch ----------------
extern "C" void kernel_launch(void* const* d_in, const int* in_sizes, int n_in,
                              void* d_out, int out_size, void* d_ws, size_t ws_size,
                              hipStream_t stream) {
  const float* emb    = (const float*)d_in[0];   // (B,N,H)
  const float* coords = (const float*)d_in[1];   // (B,N,2)
  const int*   lastn  = (const int*)  d_in[2];   // (B,G)
  const float* gmask  = (const float*)d_in[3];   // (B,G,N)
  // d_in[4] = S (unused)
  const float* Wqg    = (const float*)d_in[5];
  const float* Wqf    = (const float*)d_in[6];
  const float* Wql    = (const float*)d_in[7];
  const float* Wq     = (const float*)d_in[8];
  const float* Wvis   = (const float*)d_in[9];
  const float* Wk     = (const float*)d_in[10];
  float* out = (float*)d_out;

  char* ws = (char*)d_ws;
  size_t off = 0;
  __bf16* Kbf    = (__bf16*)(ws + off); off += (size_t)NB * NN * HDIM * 2;  // 16.38 MB
  __bf16* fqbf   = (__bf16*)(ws + off); off += (size_t)NB * NG * HDIM * 2;  //  1.64 MB
  float*  sv     = (float*) (ws + off); off += (size_t)NB * NG * HDIM * 4;  //  3.28 MB
  float*  finalq = (float*) (ws + off); off += (size_t)NB * NG * HDIM * 4;  //  3.28 MB
  float*  qg     = (float*) (ws + off); off += (size_t)NB * HDIM * 4;
  float*  Wsum   = (float*) (ws + off); off += (size_t)HDIM * HDIM * 4;
  (void)ws_size; (void)in_sizes; (void)n_in; (void)out_size;

  const int GSTRIPES = (NG + 15) / 16;   // 13

  k_wsum   <<<64, 256, 0, stream>>>(Wql, Wqf, Wsum);
  k_qgraph <<<NB, HDIM, 0, stream>>>(emb, Wqg, qg);
  k_gemm_xwt<<<(NB * NN) / 16, 256, 0, stream>>>(emb, Wk, Kbf);            // K matrix
  k_sumvis <<<dim3(GSTRIPES, NB), 256, 0, stream>>>(emb, gmask, sv);
  k_finalq <<<(NB * NG) / 16, 256, 0, stream>>>(emb, lastn, sv, Wsum, Wvis, qg, finalq);
  k_gemm_xwt<<<(NB * NG) / 16, 256, 0, stream>>>(finalq, Wq, fqbf);        // fq
  k_scores <<<dim3(16, GSTRIPES, NB), 256, 0, stream>>>(fqbf, Kbf, coords, lastn,
                                                        gmask, out);
  k_softmax<<<NB * NG, 256, 0, stream>>>(out);
}
